// ExampleWrapper_65300682769104
// MI455X (gfx1250) — matrix operand order repaired and checked by reading, not compiled, not run on hardware
//
#include <hip/hip_runtime.h>
#include <hip/hip_bf16.h>

typedef _Float16 half_t;
typedef _Float16 v16h __attribute__((ext_vector_type(16)));
typedef _Float16 h8   __attribute__((ext_vector_type(8)));
typedef float    v8f  __attribute__((ext_vector_type(8)));

#define WMMA_F16(a,b,c) __builtin_amdgcn_wmma_f32_16x16x32_f16(false,(a),false,(b),(short)0,(c),false,false)
#define CAT16(lo,hi) __builtin_shufflevector((lo),(hi),0,1,2,3,4,5,6,7,8,9,10,11,12,13,14,15)

#define TBTOK 8192
#define NSEQ  1024
#define HEADS 8
#define LAYERS 8
#define KDIM 288

// ---------------- compile-time Cl(3,0,1) tables ----------------
namespace ga {

struct Tables {
  float basis[9][16][16];
  signed char gp[16][16];   // sign*(k+1), 0 if product vanishes (e0^2=0)
  float inner[16];
};

constexpr int BL_LEN[16] = {0,1,1,1,1,2,2,2,2,2,2,3,3,3,3,4};
constexpr int BL_IDX[16][4] = {
  {0,0,0,0},{0,0,0,0},{1,0,0,0},{2,0,0,0},{3,0,0,0},
  {0,1,0,0},{0,2,0,0},{0,3,0,0},{1,2,0,0},{1,3,0,0},{2,3,0,0},
  {0,1,2,0},{0,1,3,0},{0,2,3,0},{1,2,3,0},{0,1,2,3}};

constexpr Tables make_tables() {
  Tables t{};
  signed char code[16][16] = {};
  for (int i=0;i<16;i++) {
    for (int j=0;j<16;j++) {
      int lst[8] = {}; int n=0;
      for (int u=0;u<BL_LEN[i];u++) lst[n++] = BL_IDX[i][u];
      for (int u=0;u<BL_LEN[j];u++) lst[n++] = BL_IDX[j][u];
      int sign = 1;
      for (int p=0;p<n;p++)
        for (int q=0;q+1<n;q++)
          if (lst[q] > lst[q+1]) { int tm=lst[q]; lst[q]=lst[q+1]; lst[q+1]=tm; sign=-sign; }
      int outb[8] = {}; int m=0; bool zero=false; int p=0;
      while (p<n) {
        if (p+1<n && lst[p]==lst[p+1]) { if (lst[p]==0) zero=true; p+=2; }
        else { outb[m++]=lst[p]; p+=1; }
      }
      if (zero) { code[i][j]=0; continue; }
      int ridx = 0;
      for (int bi=0;bi<16;bi++) {
        if (BL_LEN[bi]!=m) continue;
        bool eq=true;
        for (int u=0;u<m;u++) if (BL_IDX[bi][u]!=outb[u]) { eq=false; break; }
        if (eq) { ridx=bi; break; }
      }
      code[i][j] = (signed char)(sign*(ridx+1));
    }
  }
  for (int i=0;i<16;i++) for (int j=0;j<16;j++) t.gp[i][j]=code[i][j];
  for (int bI=0;bI<16;bI++) {
    bool has0=false;
    for (int u=0;u<BL_LEN[bI];u++) if (BL_IDX[bI][u]==0) has0=true;
    t.inner[bI] = has0 ? 0.0f : 1.0f;
  }
  // BASIS[0..4] = grade projections
  for (int g=0; g<5; g++)
    for (int a=0;a<16;a++)
      for (int b=0;b<16;b++)
        t.basis[g][a][b] = (a==b && BL_LEN[a]==g) ? 1.0f : 0.0f;
  // BASIS[5+g][a][b] = GP[1][b][a] if grade(b)==g
  for (int g=0; g<4; g++)
    for (int a=0;a<16;a++)
      for (int b=0;b<16;b++) {
        float v = 0.0f;
        int cd = code[1][b];
        if (cd!=0 && BL_LEN[b]==g) {
          int kk = (cd>0?cd:-cd)-1;
          if (kk==a) v = (cd>0)?1.0f:-1.0f;
        }
        t.basis[5+g][a][b] = v;
      }
  return t;
}

constexpr Tables CT = make_tables();
} // namespace ga

__constant__ ga::Tables TAB = ga::CT;

__device__ __forceinline__ float gelu_f(float v){
  float v3 = v*v*v;
  return 0.5f*v*(1.0f + tanhf(0.7978845608028654f*(v + 0.044715f*v3)));
}

// ---------------- build combined equi_linear weights (f16, M-major: Wt[m][k]) ----------------
__global__ void build_w_kernel(const float* __restrict__ wmv, const float* __restrict__ wsm,
                               const float* __restrict__ wms, const float* __restrict__ wss,
                               half_t* __restrict__ Wout, int Omv, int Os, int total) {
  int idx = blockIdx.x*blockDim.x + threadIdx.x;
  if (idx >= total) return;
  int M = Omv*16 + Os;
  int l   = idx / (M*KDIM);
  int rem = idx - l*(M*KDIM);
  int r = rem / KDIM;          // output row m
  int c = rem - r*KDIM;        // reduction index k
  const float* wmv_l = wmv + (size_t)l*Omv*16*9;
  const float* wsm_l = wsm + (size_t)l*Omv*32;
  const float* wms_l = wms + (size_t)l*Os*16;
  const float* wss_l = wss + (size_t)l*Os*32;
  float val = 0.0f;
  if (r < Omv*16) {
    int o = r >> 4, a = r & 15;
    if (c < 256) {
      int i = c >> 4, bb = c & 15;
      for (int k=0;k<9;k++) val += wmv_l[(o*16+i)*9+k] * TAB.basis[k][a][bb];
    } else if (a == 0) {
      val = wsm_l[o*32 + (c-256)];
    }
  } else {
    int o = r - Omv*16;
    if (c < 256) {
      if ((c & 15) == 0) val = wms_l[o*16 + (c>>4)];
    } else {
      val = wss_l[o*32 + (c-256)];
    }
  }
  // M-major store: row = m, col = k (contiguous in k)
  Wout[(size_t)l*KDIM*M + (size_t)r*KDIM + c] = (half_t)val;
}

__global__ void build_wout_kernel(const float* __restrict__ wout_mv,
                                  const float* __restrict__ wout_sm,
                                  float* __restrict__ wvec) {
  int c = threadIdx.x;
  if (c >= KDIM) return;
  float val = 0.0f;
  if (c < 256) {
    int i = c >> 4, bb = c & 15;
    for (int k=0;k<9;k++) val += wout_mv[i*9+k]*TAB.basis[k][0][bb];
  } else {
    val = wout_sm[c-256];
  }
  wvec[c] = val;
}

// ---------------- input embedding ----------------
__global__ void embed_kernel(const float* __restrict__ inp, const float* __restrict__ win_mv,
                             const float* __restrict__ win_bs, float* __restrict__ X, int total) {
  int t = blockIdx.x*blockDim.x + threadIdx.x;
  if (t >= total) return;
  float mv0[16] = {};
  mv0[14] = 1.0f;
  mv0[13] = -inp[t*3+0];
  mv0[12] =  inp[t*3+1];
  mv0[11] = -inp[t*3+2];
  float b0[9][16];
  for (int k=0;k<9;k++)
    for (int a=0;a<16;a++){
      float s = 0.0f;
      for (int bb=11;bb<=14;bb++) s += TAB.basis[k][a][bb]*mv0[bb];
      b0[k][a]=s;
    }
  float* xt = X + (size_t)t*KDIM;
  for (int o=0;o<16;o++)
    for (int a=0;a<16;a++){
      float s = 0.0f;
      for (int k=0;k<9;k++) s += win_mv[o*9+k]*b0[k][a];
      xt[o*16+a]=s;
    }
  for (int j=0;j<32;j++) xt[256+j]=win_bs[j];   // mv0[...,0]==0 -> s = bias
}

// ---------------- equivariant layernorm -> f16 ----------------
__global__ void ln_kernel(const float* __restrict__ X, half_t* __restrict__ Xn, int total) {
  int t = blockIdx.x*blockDim.x + threadIdx.x;
  if (t >= total) return;
  const float* xt = X + (size_t)t*KDIM;
  float f = 0.0f;
  for (int c=0;c<16;c++) {
    #pragma unroll
    for (int bb=0;bb<16;bb++){ float v = xt[c*16+bb]; f += v*v*ga::CT.inner[bb]; }
  }
  float rmv = rsqrtf(f*(1.0f/16.0f) + 1e-6f);
  float ss = 0.0f;
  for (int j=0;j<32;j++){ float v = xt[256+j]; ss += v*v; }
  float rs = rsqrtf(ss*(1.0f/32.0f) + 1e-6f);
  half_t* xo = Xn + (size_t)t*KDIM;
  for (int i=0;i<256;i++) xo[i] = (half_t)(xt[i]*rmv);
  for (int j=0;j<32;j++)  xo[256+j] = (half_t)(xt[256+j]*rs);
}

// ---------------- token GEMM: Y[T,M] = X[T,288] * Wt[288,M]; one wave = 16x32 tile ----------------
__global__ void __launch_bounds__(128)
gemm_kernel(const half_t* __restrict__ X, const half_t* __restrict__ Wt,
            float* __restrict__ Y, int M, int tiles_m2, int tiles_total) {
  int wave = blockIdx.x * 4 + (threadIdx.x >> 5);
  if (wave >= tiles_total) return;           // wave-uniform; EXEC stays all-ones for WMMA
  int lane = threadIdx.x & 31;
  int tt = wave / tiles_m2;
  int mt = wave - tt*tiles_m2;
  int tb = tt << 4, mb = mt << 5;
  int row = lane & 15, grp = lane >> 4;
  v8f acc0 = {}, acc1 = {};
  const half_t* xr0 = X  + (size_t)(tb + row)*KDIM + grp*8;
  const half_t* wr0 = Wt + (size_t)(mb + row)*KDIM + grp*8;   // row = output column n
  #pragma unroll
  for (int kb=0; kb<KDIM; kb+=32) {
    // A (16x32): lane-half grp -> K = kb + grp*8 + {0..7} and +16 (two contiguous h8 runs)
    h8 lo = *(const h8*)(xr0 + kb);
    h8 hi = *(const h8*)(xr0 + kb + 16);
    v16h a = CAT16(lo, hi);
    // B (32x16)x2: lane-half grp -> column n, K = kb + grp*8 + {0..7} and +16 (M-major weights)
    const half_t* wp = wr0 + kb;
    if (kb + 64 < KDIM) __builtin_prefetch(wp + 64, 0, 0);
    h8 w0 = *(const h8*)(wp);
    h8 w1 = *(const h8*)(wp+16);
    h8 w2 = *(const h8*)(wp+(size_t)16*KDIM);
    h8 w3 = *(const h8*)(wp+(size_t)16*KDIM+16);
    v16h b0 = CAT16(w0, w1);
    v16h b1 = CAT16(w2, w3);
    acc0 = WMMA_F16(a, b0, acc0);
    acc1 = WMMA_F16(a, b1, acc1);
  }
  // D: lane = col, VGPR r -> row r + 8*grp
  float* yp = Y + (size_t)tb*M + mb + row;
  #pragma unroll
  for (int r=0;r<8;r++) {
    yp[(size_t)(r + grp*8)*M]      = acc0[r];
    yp[(size_t)(r + grp*8)*M + 16] = acc1[r];
  }
}

// ---------------- split qkv into padded per-head Q/K^T/V (f16) ----------------
// Q: [bh][n][64]  (row-major, A-fragment friendly)
// K: [bh][n][64]  (B-fragment rows contiguous over dims)
// V: [bh][48][n]  (transposed -> B-fragment rows contiguous over keys)
__global__ void prep_qkv_kernel(const float* __restrict__ Y, half_t* __restrict__ Q,
                                half_t* __restrict__ Kt, half_t* __restrict__ Vb, int total) {
  int idx = blockIdx.x*blockDim.x + threadIdx.x;   // over BH*N
  if (idx >= total) return;
  int bh = idx / NSEQ, n = idx - bh*NSEQ;
  int b = bh >> 3, h = bh & 7;
  const float* y = Y + (size_t)(b*NSEQ+n)*864;
  const float SCALE = 0.22360679774997896f;        // 1/sqrt(8*(C/H) + S/H) = 1/sqrt(20)
  half_t* q  = Q  + (size_t)idx*64;
  half_t* kc = Kt + (size_t)idx*64;
  half_t* v  = Vb + (size_t)bh*48*NSEQ + n;        // column n of V^T
  for (int d=0; d<64; d++) {
    float qv = 0.0f, kv = 0.0f;
    if (d < 32) {
      int cc = d>>4, a = d&15;
      qv = y[(h*2+cc)*16 + a] * TAB.inner[a];
      kv = y[(16 + h*2+cc)*16 + a];
    } else if (d < 36) {
      qv = y[768      + h*4 + (d-32)];
      kv = y[768 + 32 + h*4 + (d-32)];
    }
    q[d] = (half_t)(qv*SCALE);
    kc[d] = (half_t)kv;
  }
  for (int d=0; d<48; d++) {
    float vv = 0.0f;
    if (d < 32) { int cc = d>>4, a = d&15; vv = y[(32 + h*2+cc)*16 + a]; }
    else if (d < 36) vv = y[768 + 64 + h*4 + (d-32)];
    v[(size_t)d*NSEQ] = (half_t)vv;
  }
}

// ---------------- flash attention: 1 wave per (head, 16-query tile) ----------------
__global__ void __launch_bounds__(32)
flash_kernel(const half_t* __restrict__ Q, const half_t* __restrict__ Ktg,
             const half_t* __restrict__ Vg, half_t* __restrict__ AO) {
  __shared__ __align__(16) half_t Pt[16][32];
  int qt = blockIdx.x & 63;
  int bh = blockIdx.x >> 6;
  int b = bh >> 3, h = bh & 7;
  int lane = threadIdx.x;
  int row = lane & 15, grp = lane >> 4;
  int qbase = qt * 16;
  const half_t* Qb = Q   + (size_t)bh*NSEQ*64;
  const half_t* Kb = Ktg + (size_t)bh*64*NSEQ;   // [n][dim]
  const half_t* Vb = Vg  + (size_t)bh*NSEQ*48;   // [dim][n]
  // Q A-fragments for d-chunks 0 and 1
  v16h qa0, qa1;
  {
    const half_t* qr = Qb + (size_t)(qbase+row)*64 + grp*8;
    qa0 = CAT16(*(const h8*)qr,      *(const h8*)(qr+16));
    qa1 = CAT16(*(const h8*)(qr+32), *(const h8*)(qr+48));
  }
  v8f o0={}, o1={}, o2={};
  float mi[8], li[8];
  #pragma unroll
  for (int r=0;r<8;r++){ mi[r] = -1e30f; li[r] = 0.0f; }

  for (int kb=0; kb<NSEQ; kb+=32) {
    // S = Q K^T : B lane = key, elements = dims grp*8+{0..7} and +16 straight from global
    const half_t* kr0 = Kb + (size_t)(kb+row)*64 + grp*8;
    const half_t* kr1 = kr0 + 16*64;
    h8 a0 = *(const h8*)(kr0);
    h8 a1 = *(const h8*)(kr0 + 16);
    h8 a2 = *(const h8*)(kr1);
    h8 a3 = *(const h8*)(kr1 + 16);
    h8 c0 = *(const h8*)(kr0 + 32);
    h8 c1 = *(const h8*)(kr0 + 48);
    h8 c2 = *(const h8*)(kr1 + 32);
    h8 c3 = *(const h8*)(kr1 + 48);
    v16h b00 = CAT16(a0, a1);   // keys  0-15, dims  0-31
    v16h b10 = CAT16(a2, a3);   // keys 16-31, dims  0-31
    v16h b01 = CAT16(c0, c1);   // keys  0-15, dims 32-63
    v16h b11 = CAT16(c2, c3);   // keys 16-31, dims 32-63
    v8f s0={}, s1={};
    s0 = WMMA_F16(qa0, b00, s0);
    s0 = WMMA_F16(qa1, b01, s0);
    s1 = WMMA_F16(qa0, b10, s1);
    s1 = WMMA_F16(qa1, b11, s1);

    // online softmax; rows r+8*grp live across the 16 lanes of each half
    #pragma unroll
    for (int r=0;r<8;r++){
      float mx = fmaxf(s0[r], s1[r]);
      mx = fmaxf(mx, __shfl_xor(mx, 1, 32));
      mx = fmaxf(mx, __shfl_xor(mx, 2, 32));
      mx = fmaxf(mx, __shfl_xor(mx, 4, 32));
      mx = fmaxf(mx, __shfl_xor(mx, 8, 32));
      float mn = fmaxf(mi[r], mx);
      float alpha = __expf(mi[r] - mn);
      float p0 = __expf(s0[r] - mn);
      float p1 = __expf(s1[r] - mn);
      float rs = p0 + p1;
      rs += __shfl_xor(rs, 1, 32);
      rs += __shfl_xor(rs, 2, 32);
      rs += __shfl_xor(rs, 4, 32);
      rs += __shfl_xor(rs, 8, 32);
      li[r] = li[r]*alpha + rs;
      mi[r] = mn;
      o0[r] *= alpha; o1[r] *= alpha; o2[r] *= alpha;
      Pt[r + grp*8][row]    = (half_t)p0;   // D-layout -> LDS transpose staging
      Pt[r + grp*8][16+row] = (half_t)p1;
    }
    __syncthreads();

    // O += P V : A from Pt (two contiguous h8 runs), B straight from global V
    const half_t* pr = &Pt[row][grp*8];
    v16h pa = CAT16(*(const h8*)pr, *(const h8*)(pr+16));
    const half_t* vp = Vb + (size_t)row*NSEQ + kb + grp*8;   // lane = dim
    v16h vb0 = CAT16(*(const h8*)(vp),      *(const h8*)(vp+16));
    v16h vb1 = CAT16(*(const h8*)(vp+(size_t)16*NSEQ),   *(const h8*)(vp+(size_t)16*NSEQ+16));
    v16h vb2 = CAT16(*(const h8*)(vp+(size_t)32*NSEQ),   *(const h8*)(vp+(size_t)32*NSEQ+16));
    o0 = WMMA_F16(pa, vb0, o0);
    o1 = WMMA_F16(pa, vb1, o1);
    o2 = WMMA_F16(pa, vb2, o2);
    __syncthreads();   // Pt rewritten next iteration
  }

  // epilogue: normalize and scatter heads back into [T,288] f16 layout
  #pragma unroll
  for (int r=0;r<8;r++){
    float inv = 1.0f / li[r];
    int qrow = r + grp*8;
    half_t* ao = AO + (size_t)(b*NSEQ + qbase + qrow)*KDIM;
    ao[h*32 + row]      = (half_t)(o0[r]*inv);
    ao[h*32 + 16 + row] = (half_t)(o1[r]*inv);
    if (row < 4) ao[256 + h*4 + row] = (half_t)(o2[r]*inv);
  }
}

// ---------------- geometric product + gating ----------------
__global__ void geoprod_mv_kernel(const float* __restrict__ Hb, half_t* __restrict__ G, int total) {
  int idx = blockIdx.x*blockDim.x + threadIdx.x;
  if (idx >= total) return;
  int t = idx >> 4, c = idx & 15;
  const float* hb = Hb + (size_t)t*576;
  float a1[16], a2[16];
  #pragma unroll
  for (int i=0;i<16;i++){ a1[i] = hb[c*16+i]; a2[i] = hb[256 + c*16 + i]; }
  float outv[16] = {};
  #pragma unroll
  for (int i=0;i<16;i++){
    #pragma unroll
    for (int j=0;j<16;j++){
      const int cd = ga::CT.gp[i][j];
      if (cd > 0)      outv[ cd-1] += a1[i]*a2[j];
      else if (cd < 0) outv[-cd-1] -= a1[i]*a2[j];
    }
  }
  float g = gelu_f(outv[0]);
  half_t* go = G + (size_t)t*KDIM + c*16;
  #pragma unroll
  for (int k=0;k<16;k++) go[k] = (half_t)(outv[k]*g);
}

__global__ void geoprod_s_kernel(const float* __restrict__ Hb, half_t* __restrict__ G, int total) {
  int idx = blockIdx.x*blockDim.x + threadIdx.x;
  if (idx >= total) return;
  int t = idx >> 5, j = idx & 31;
  float v = Hb[(size_t)t*576 + 512 + j] * gelu_f(Hb[(size_t)t*576 + 544 + j]);
  G[(size_t)t*KDIM + 256 + j] = (half_t)v;
}

// ---------------- residual add ----------------
__global__ void resadd_kernel(float* __restrict__ X, const float* __restrict__ D, int total) {
  int i = blockIdx.x*blockDim.x + threadIdx.x;
  if (i < total) X[i] += D[i];
}

// ---------------- final projection + deterministic mean over tokens ----------------
__global__ void final_kernel(const float* __restrict__ X, const float* __restrict__ wvec,
                             float* __restrict__ out) {
  __shared__ float red[256];
  int b = blockIdx.x;
  float acc = 0.0f;
  for (int n = threadIdx.x; n < NSEQ; n += 256) {
    const float* xt = X + (size_t)(b*NSEQ + n)*KDIM;
    float d = 0.0f;
    for (int i=0;i<KDIM;i++) d += xt[i]*wvec[i];
    acc += d;
  }
  red[threadIdx.x] = acc;
  __syncthreads();
  for (int s2=128; s2>0; s2>>=1){
    if (threadIdx.x < s2) red[threadIdx.x] += red[threadIdx.x + s2];
    __syncthreads();
  }
  if (threadIdx.x == 0) out[b] = red[0] * (1.0f/(float)NSEQ);
}

// ---------------- host orchestration ----------------
extern "C" void kernel_launch(void* const* d_in, const int* in_sizes, int n_in,
                              void* d_out, int out_size, void* d_ws, size_t ws_size,
                              hipStream_t stream) {
  (void)in_sizes; (void)n_in; (void)out_size; (void)ws_size;
  const float* inp      = (const float*)d_in[0];
  const float* win_mv   = (const float*)d_in[1];
  /* d_in[2] win_ms multiplies mv0[...,0]==0 -> exactly zero, unused */
  const float* win_bs   = (const float*)d_in[3];
  const float* qkv_wmv  = (const float*)d_in[4];
  const float* qkv_wsm  = (const float*)d_in[5];
  const float* qkv_wms  = (const float*)d_in[6];
  const float* qkv_wss  = (const float*)d_in[7];
  const float* ao_wmv   = (const float*)d_in[8];
  const float* ao_wsm   = (const float*)d_in[9];
  const float* ao_wms   = (const float*)d_in[10];
  const float* ao_wss   = (const float*)d_in[11];
  const float* m1_wmv   = (const float*)d_in[12];
  const float* m1_wsm   = (const float*)d_in[13];
  const float* m1_wms   = (const float*)d_in[14];
  const float* m1_wss   = (const float*)d_in[15];
  const float* m2_wmv   = (const float*)d_in[16];
  const float* m2_wsm   = (const float*)d_in[17];
  const float* m2_wms   = (const float*)d_in[18];
  const float* m2_wss   = (const float*)d_in[19];
  const float* wout_mv  = (const float*)d_in[20];
  const float* wout_sm  = (const float*)d_in[21];

  char* base = (char*)d_ws;
  size_t off = 0;
  auto alloc = [&](size_t bytes) -> char* {
    char* p = base + off;
    off += (bytes + 255) & ~(size_t)255;
    return p;
  };
  float*  x    = (float*) alloc((size_t)TBTOK*KDIM*4);
  float*  ybig = (float*) alloc((size_t)TBTOK*864*4);
  float*  dbuf = (float*) alloc((size_t)TBTOK*KDIM*4);
  float*  wvec = (float*) alloc(KDIM*4);
  half_t* xn   = (half_t*)alloc((size_t)TBTOK*KDIM*2);
  half_t* ao   = (half_t*)alloc((size_t)TBTOK*KDIM*2);
  half_t* Qb   = (half_t*)alloc((size_t)64*NSEQ*64*2);
  half_t* Kb   = (half_t*)alloc((size_t)64*NSEQ*64*2);
  half_t* Vb   = (half_t*)alloc((size_t)64*NSEQ*48*2);
  half_t* Wqkv = (half_t*)alloc((size_t)LAYERS*864*KDIM*2);
  half_t* Wao  = (half_t*)alloc((size_t)LAYERS*288*KDIM*2);
  half_t* Wm1  = (half_t*)alloc((size_t)LAYERS*576*KDIM*2);
  half_t* Wm2  = (half_t*)alloc((size_t)LAYERS*288*KDIM*2);

  // fold BASIS into per-layer combined weights (once per call)
  { int tot = LAYERS*864*KDIM;
    build_w_kernel<<<(tot+255)/256,256,0,stream>>>(qkv_wmv,qkv_wsm,qkv_wms,qkv_wss,Wqkv,48,96,tot); }
  { int tot = LAYERS*288*KDIM;
    build_w_kernel<<<(tot+255)/256,256,0,stream>>>(ao_wmv,ao_wsm,ao_wms,ao_wss,Wao,16,32,tot); }
  { int tot = LAYERS*576*KDIM;
    build_w_kernel<<<(tot+255)/256,256,0,stream>>>(m1_wmv,m1_wsm,m1_wms,m1_wss,Wm1,32,64,tot); }
  { int tot = LAYERS*288*KDIM;
    build_w_kernel<<<(tot+255)/256,256,0,stream>>>(m2_wmv,m2_wsm,m2_wms,m2_wss,Wm2,16,32,tot); }
  build_wout_kernel<<<1,KDIM,0,stream>>>(wout_mv, wout_sm, wvec);

  embed_kernel<<<(TBTOK+255)/256,256,0,stream>>>(inp, win_mv, win_bs, x, TBTOK);

  for (int l=0; l<LAYERS; l++) {
    // --- attention block ---
    ln_kernel<<<(TBTOK+255)/256,256,0,stream>>>(x, xn, TBTOK);
    { int tm2 = 864/32, tot = (TBTOK/16)*tm2;
      gemm_kernel<<<(tot+3)/4,128,0,stream>>>(xn, Wqkv + (size_t)l*864*KDIM, ybig, 864, tm2, tot); }
    prep_qkv_kernel<<<(64*NSEQ+255)/256,256,0,stream>>>(ybig, Qb, Kb, Vb, 64*NSEQ);
    flash_kernel<<<64*64,32,0,stream>>>(Qb, Kb, Vb, ao);
    { int tm2 = 288/32, tot = (TBTOK/16)*tm2;
      gemm_kernel<<<(tot+3)/4,128,0,stream>>>(ao, Wao + (size_t)l*288*KDIM, dbuf, 288, tm2, tot); }
    resadd_kernel<<<(TBTOK*KDIM+255)/256,256,0,stream>>>(x, dbuf, TBTOK*KDIM);
    // --- geometric MLP block ---
    ln_kernel<<<(TBTOK+255)/256,256,0,stream>>>(x, xn, TBTOK);
    { int tm2 = 576/32, tot = (TBTOK/16)*tm2;
      gemm_kernel<<<(tot+3)/4,128,0,stream>>>(xn, Wm1 + (size_t)l*576*KDIM, ybig, 576, tm2, tot); }
    geoprod_mv_kernel<<<(TBTOK*16+255)/256,256,0,stream>>>(ybig, ao, TBTOK*16);
    geoprod_s_kernel <<<(TBTOK*32+255)/256,256,0,stream>>>(ybig, ao, TBTOK*32);
    { int tm2 = 288/32, tot = (TBTOK/16)*tm2;
      gemm_kernel<<<(tot+3)/4,128,0,stream>>>(ao, Wm2 + (size_t)l*288*KDIM, dbuf, 288, tm2, tot); }
    resadd_kernel<<<(TBTOK*KDIM+255)/256,256,0,stream>>>(x, dbuf, TBTOK*KDIM);
  }

  final_kernel<<<8,256,0,stream>>>(x, wvec, (float*)d_out);
}
